// GatV2Classification_84576495993385
// MI455X (gfx1250) — matrix operand order repaired and checked by reading, not compiled, not run on hardware
//
#include <hip/hip_runtime.h>
#include <hip/hip_bf16.h>

#define NNODES 40000
#define NEDGES 640000
#define DIM    256
#define NEG    0.2f

typedef __attribute__((ext_vector_type(16))) __bf16 v16bf;
typedef __attribute__((ext_vector_type(8)))  float  v8f;

// ---------------------------------------------------------------------------
// GEMM: O0 = X @ W0 + b0 and O1 = X @ W1 + b1 computed together.
// One wave32 computes a 16(M) x 64(N) tile for BOTH weight matrices:
// one A fragment (bf16, built once per K-step) feeds 8 WMMAs
// (4 N-tiles x 2 weights), maximizing register reuse of X.
// v_wmma_f32_16x16x32_bf16, f32 accumulate, K looped in steps of 32.
//
// Fragment layouts used here (wave32, contiguous k order in A and B):
//   A (16x32 bf16): lane L holds row M = L&15; element j holds
//                   K = (L>>4)*16 + j
//   B (32x16 bf16): lane L holds col N = L&15; element j holds
//                   K = (L>>4)*16 + j
//   C/D (16x16 f32): vgpr r, lane L -> M = (L>>4)*8 + r, N = L&15
// ---------------------------------------------------------------------------
__global__ __launch_bounds__(32)
void gemm_bf16_wmma(const float* __restrict__ X,
                    const float* __restrict__ W0, const float* __restrict__ W1,
                    const float* __restrict__ b0, const float* __restrict__ b1,
                    float* __restrict__ O0, float* __restrict__ O1)
{
    const int lane = threadIdx.x;
    const int half = lane >> 4;
    const int l16  = lane & 15;
    const int m0 = blockIdx.x * 16;
    const int n0 = blockIdx.y * 64;

    v8f acc[2][4] = {};
    const float* xrow_base = X + (size_t)(m0 + l16) * DIM;

    for (int k0 = 0; k0 < DIM; k0 += 32) {
        // A fragment: built once, reused by all 8 WMMAs this K-step.
        v16bf a;
        const float* xrow = xrow_base + k0 + half * 16;
        #pragma unroll
        for (int j = 0; j < 8; ++j) a[j]     = (__bf16)xrow[j];
        #pragma unroll
        for (int j = 0; j < 8; ++j) a[8 + j] = (__bf16)xrow[8 + j];

        #pragma unroll
        for (int w = 0; w < 2; ++w) {
            const float* W = w ? W1 : W0;
            #pragma unroll
            for (int t = 0; t < 4; ++t) {
                v16bf bb;
                const float* wcol =
                    W + (size_t)(k0 + half * 16) * DIM + n0 + t * 16 + l16;
                #pragma unroll
                for (int j = 0; j < 16; ++j) bb[j] = (__bf16)wcol[(size_t)j * DIM];
                acc[w][t] = __builtin_amdgcn_wmma_f32_16x16x32_bf16(
                    false, a, false, bb, (short)0, acc[w][t], false, false);
            }
        }
    }

    #pragma unroll
    for (int w = 0; w < 2; ++w) {
        const float* b = w ? b1 : b0;
        float*       O = w ? O1 : O0;
        #pragma unroll
        for (int t = 0; t < 4; ++t) {
            const float bias = b[n0 + t * 16 + l16];
            #pragma unroll
            for (int r = 0; r < 8; ++r) {
                const int m = m0 + half * 8 + r;
                O[(size_t)m * DIM + n0 + t * 16 + l16] = acc[w][t][r] + bias;
            }
        }
    }
}

// ---------------------------------------------------------------------------
// Monotone float <-> u32 encoding so atomicMax(u32) orders like float.
// ---------------------------------------------------------------------------
__device__ __forceinline__ unsigned enc_f32(float f) {
    unsigned u = __float_as_uint(f);
    return (u & 0x80000000u) ? ~u : (u | 0x80000000u);
}
__device__ __forceinline__ float dec_f32(unsigned e) {
    unsigned u = (e & 0x80000000u) ? (e ^ 0x80000000u) : ~e;
    return __uint_as_float(u);
}

__global__ void fill_f32(float* __restrict__ p, float v, int n) {
    int i = blockIdx.x * blockDim.x + threadIdx.x;
    if (i < n) p[i] = v;
}
__global__ void fill_u32(unsigned* __restrict__ p, unsigned v, int n) {
    int i = blockIdx.x * blockDim.x + threadIdx.x;
    if (i < n) p[i] = v;
}

// ---------------------------------------------------------------------------
// Pass 1: logit[e] = att . leaky_relu(xl[src] + xr[dst]); segment-max(dst).
// One wave32 per edge; each lane covers 8 dims via two float4 loads.
// Self-loop edges are e in [E, E+N): src = dst = e - E.
// ---------------------------------------------------------------------------
__global__ __launch_bounds__(256)
void edge_logits(const float* __restrict__ xl, const float* __restrict__ xr,
                 const float* __restrict__ att,
                 const int* __restrict__ src, const int* __restrict__ dst,
                 int nGraphEdges, int nTotal,
                 float* __restrict__ logit, unsigned* __restrict__ maxenc)
{
    const int e = blockIdx.x * (blockDim.x >> 5) + (threadIdx.x >> 5);
    if (e >= nTotal) return;
    const int lane = threadIdx.x & 31;

    int s, d;
    if (e < nGraphEdges) { s = src[e]; d = dst[e]; }
    else                 { s = e - nGraphEdges; d = s; }

    const float4* pl = (const float4*)(xl + (size_t)s * DIM) + lane * 2;
    const float4* pr = (const float4*)(xr + (size_t)d * DIM) + lane * 2;
    const float4* pa = (const float4*)att + lane * 2;

    float acc = 0.f;
    #pragma unroll
    for (int q = 0; q < 2; ++q) {
        float4 a = pl[q], r = pr[q], t = pa[q];
        float v;
        v = a.x + r.x; v = (v > 0.f) ? v : NEG * v; acc += v * t.x;
        v = a.y + r.y; v = (v > 0.f) ? v : NEG * v; acc += v * t.y;
        v = a.z + r.z; v = (v > 0.f) ? v : NEG * v; acc += v * t.z;
        v = a.w + r.w; v = (v > 0.f) ? v : NEG * v; acc += v * t.w;
    }
    #pragma unroll
    for (int off = 16; off; off >>= 1) acc += __shfl_xor(acc, off, 32);

    if (lane == 0) {
        logit[e] = acc;
        atomicMax(maxenc + d, enc_f32(acc));
    }
}

// Pass 2: ex[e] = exp(logit[e] - max[dst]); segment-sum(dst). In-place over logit.
__global__ __launch_bounds__(256)
void edge_exp(float* __restrict__ elog, const int* __restrict__ dst,
              int nGraphEdges, int nTotal,
              const unsigned* __restrict__ maxenc, float* __restrict__ ssum)
{
    const int e = blockIdx.x * blockDim.x + threadIdx.x;
    if (e >= nTotal) return;
    const int d = (e < nGraphEdges) ? dst[e] : (e - nGraphEdges);
    const float v = __expf(elog[e] - dec_f32(maxenc[d]));
    elog[e] = v;
    atomicAdd(ssum + d, v);
}

// Pass 3: acc[dst] += ex[e] * xl[src]. One wave32 per edge, 8 dims/lane.
__global__ __launch_bounds__(256)
void edge_aggregate(const float* __restrict__ xl, const float* __restrict__ ex,
                    const int* __restrict__ src, const int* __restrict__ dst,
                    int nGraphEdges, int nTotal, float* __restrict__ acc)
{
    const int e = blockIdx.x * (blockDim.x >> 5) + (threadIdx.x >> 5);
    if (e >= nTotal) return;
    const int lane = threadIdx.x & 31;

    int s, d;
    if (e < nGraphEdges) { s = src[e]; d = dst[e]; }
    else                 { s = e - nGraphEdges; d = s; }

    const float a = ex[e];
    const float* xs = xl + (size_t)s * DIM + lane * 8;
    float*       ad = acc + (size_t)d * DIM + lane * 8;
    #pragma unroll
    for (int i = 0; i < 8; ++i) atomicAdd(ad + i, a * xs[i]);
}

// Pass 4: y = leaky_relu(acc / sum[node] + bias[c])  (in-place over acc ok).
__global__ __launch_bounds__(256)
void node_finalize(const float* __restrict__ acc, const float* __restrict__ ssum,
                   const float* __restrict__ bias, float* __restrict__ y, int n)
{
    const int idx = blockIdx.x * blockDim.x + threadIdx.x;
    if (idx >= n * DIM) return;
    const int node = idx >> 8;
    const int c    = idx & (DIM - 1);
    float v = acc[idx] / ssum[node] + bias[c];
    y[idx] = (v > 0.f) ? v : NEG * v;
}

// Head: z = y @ W_lin + b_lin (256 -> 2), log_softmax. One wave32 per node.
__global__ __launch_bounds__(256)
void classifier(const float* __restrict__ y, const float* __restrict__ Wlin,
                const float* __restrict__ blin, float* __restrict__ out, int n)
{
    const int node = blockIdx.x * (blockDim.x >> 5) + (threadIdx.x >> 5);
    if (node >= n) return;
    const int lane = threadIdx.x & 31;

    const float* yr = y + (size_t)node * DIM + lane * 8;
    float z0 = 0.f, z1 = 0.f;
    #pragma unroll
    for (int i = 0; i < 8; ++i) {
        const float v = yr[i];
        const int c = lane * 8 + i;
        z0 += v * Wlin[c * 2 + 0];
        z1 += v * Wlin[c * 2 + 1];
    }
    #pragma unroll
    for (int off = 16; off; off >>= 1) {
        z0 += __shfl_xor(z0, off, 32);
        z1 += __shfl_xor(z1, off, 32);
    }
    if (lane == 0) {
        z0 += blin[0]; z1 += blin[1];
        const float m = fmaxf(z0, z1);
        const float lse = m + __logf(__expf(z0 - m) + __expf(z1 - m));
        out[node * 2 + 0] = z0 - lse;
        out[node * 2 + 1] = z1 - lse;
    }
}

// ---------------------------------------------------------------------------
extern "C" void kernel_launch(void* const* d_in, const int* in_sizes, int n_in,
                              void* d_out, int out_size, void* d_ws, size_t ws_size,
                              hipStream_t stream)
{
    (void)in_sizes; (void)n_in; (void)out_size; (void)ws_size;

    const float* fts  = (const float*)d_in[0];
    const int*   g    = (const int*)  d_in[1];
    const float* Wl0  = (const float*)d_in[2];
    const float* Wr0  = (const float*)d_in[3];
    const float* bl0  = (const float*)d_in[4];
    const float* br0  = (const float*)d_in[5];
    const float* att0 = (const float*)d_in[6];
    const float* b0   = (const float*)d_in[7];
    const float* Wl1  = (const float*)d_in[8];
    const float* Wr1  = (const float*)d_in[9];
    const float* bl1  = (const float*)d_in[10];
    const float* br1  = (const float*)d_in[11];
    const float* att1 = (const float*)d_in[12];
    const float* b1   = (const float*)d_in[13];
    const float* Wlin = (const float*)d_in[14];
    const float* blin = (const float*)d_in[15];

    const int T = 4, N = NNODES, E = NEDGES, ET = E + N;

    // Workspace layout (f32 slots; acc/y buffer reused across both layers).
    float*    xl     = (float*)d_ws;                 // N*256
    float*    xr     = xl   + (size_t)N * DIM;       // N*256
    float*    accY   = xr   + (size_t)N * DIM;       // N*256  (acc -> y, reused)
    float*    elog   = accY + (size_t)N * DIM;       // ET     (logits -> ex, in place)
    float*    ssum   = elog + ET;                    // N
    unsigned* maxenc = (unsigned*)(ssum + N);        // N

    const int* src0 = g;                               // graph[0,0,:]
    const int* dst0 = g + E;                           // graph[0,1,:]
    const int* srcL = g + (size_t)(T - 1) * 2 * E;     // graph[T-1,0,:]
    const int* dstL = srcL + E;                        // graph[T-1,1,:]

    auto run_layer = [&](const float* xin, const int* src, const int* dst,
                         const float* Wl, const float* Wr,
                         const float* bl, const float* br,
                         const float* att, const float* bias)
    {
        gemm_bf16_wmma<<<dim3(N / 16, DIM / 64), 32, 0, stream>>>(
            xin, Wl, Wr, bl, br, xl, xr);
        fill_u32<<<(N + 255) / 256, 256, 0, stream>>>(maxenc, 0u, N);
        fill_f32<<<(N + 255) / 256, 256, 0, stream>>>(ssum, 0.f, N);
        edge_logits<<<(ET + 7) / 8, 256, 0, stream>>>(
            xl, xr, att, src, dst, E, ET, elog, maxenc);
        edge_exp<<<(ET + 255) / 256, 256, 0, stream>>>(
            elog, dst, E, ET, maxenc, ssum);
        fill_f32<<<(N * DIM + 255) / 256, 256, 0, stream>>>(accY, 0.f, N * DIM);
        edge_aggregate<<<(ET + 7) / 8, 256, 0, stream>>>(
            xl, elog, src, dst, E, ET, accY);
        node_finalize<<<(N * DIM + 255) / 256, 256, 0, stream>>>(
            accY, ssum, bias, accY, N);
    };

    // Layer 0: input fts[0], edges graph[0] (+ self loops).
    run_layer(fts, src0, dst0, Wl0, Wr0, bl0, br0, att0, b0);
    // Layer 1: input y0 (accY), edges graph[T-1] (+ self loops).
    run_layer(accY, srcL, dstL, Wl1, Wr1, bl1, br1, att1, b1);
    // Head + log-softmax.
    classifier<<<(N + 7) / 8, 256, 0, stream>>>(accY, Wlin, blin, (float*)d_out, N);
}
